// TransformerBlock_16088947491310
// MI455X (gfx1250) — matrix operand order repaired and checked by reading, not compiled, not run on hardware
//
#include <hip/hip_runtime.h>
#include <math.h>

typedef __bf16 bf16;
typedef __attribute__((ext_vector_type(16))) __bf16 v16bf;
typedef __attribute__((ext_vector_type(8)))  __bf16 v8bf;
typedef __attribute__((ext_vector_type(8)))  float  v8f;

typedef int v4i __attribute__((vector_size(16)));
typedef __attribute__((address_space(1))) v4i as1_v4i;
typedef __attribute__((address_space(3))) v4i as3_v4i;

#define WMMA_BF16(a, b, c) \
  __builtin_amdgcn_wmma_f32_16x16x32_bf16(false, (a), false, (b), (short)0, (c), false, false)

// ---- CDNA5 async global->LDS copy (ASYNCcnt path), 16 bytes per lane --------
__device__ __forceinline__ void async_cp16(void* lds_dst, const void* gsrc) {
#if __has_builtin(__builtin_amdgcn_global_load_async_to_lds_b128)
  __builtin_amdgcn_global_load_async_to_lds_b128(
      (as1_v4i*)gsrc, (as3_v4i*)lds_dst, 0, 0);
#else
  asm volatile("global_load_async_to_lds_b128 %0, %1, off"
               :: "v"((unsigned)(unsigned long long)lds_dst),
                  "v"((unsigned long long)gsrc)
               : "memory");
#endif
}
__device__ __forceinline__ void async_wait0() {
#if __has_builtin(__builtin_amdgcn_s_wait_asynccnt)
  __builtin_amdgcn_s_wait_asynccnt(0);
#else
  asm volatile("s_wait_asynccnt 0x0" ::: "memory");
#endif
}

__device__ __forceinline__ v8f zero8() {
  v8f z;
#pragma unroll
  for (int e = 0; e < 8; ++e) z[e] = 0.f;
  return z;
}

// A-fragment (16x32 bf16): lanes 0-15 = rows, K per ISA layout:
// lane half hf: K = hf*8 + e (e<8) and 16 + hf*8 + (e-8)  -> two 16B runs.
__device__ __forceinline__ v16bf frag_A(const bf16* p, int ld) {
  int lane = threadIdx.x & 31;
  const bf16* q = p + (lane & 15) * ld + (lane >> 4) * 8;
  v8bf lo = *(const v8bf*)q;
  v8bf hi = *(const v8bf*)(q + 16);
  v16bf o;
#pragma unroll
  for (int e = 0; e < 8; ++e) { o[e] = lo[e]; o[e + 8] = hi[e]; }
  return o;
}

// B-fragment (32x16 bf16) read from an N-major (transposed) tile:
// lane: n = lane&15, K = (lane>>4)*8 + e (e<8) and 16 + (lane>>4)*8 + (e-8).
__device__ __forceinline__ v16bf frag_B(const bf16* p, int ld) {
  int lane = threadIdx.x & 31;
  const bf16* q = p + (lane & 15) * ld + (lane >> 4) * 8;
  v8bf lo = *(const v8bf*)q;
  v8bf hi = *(const v8bf*)(q + 16);
  v16bf o;
#pragma unroll
  for (int e = 0; e < 8; ++e) { o[e] = lo[e]; o[e + 8] = hi[e]; }
  return o;
}

// ---------------- weight convert + transpose: WT[n*K+k] = (bf16)W[k*N+n] -----
__global__ __launch_bounds__(256) void wt_kernel(const float* __restrict__ W,
                                                 bf16* __restrict__ WT,
                                                 int K, int N) {
  size_t idx = (size_t)blockIdx.x * 256 + threadIdx.x;
  if (idx >= (size_t)K * N) return;
  size_t n = idx / K, k = idx % K;
  WT[idx] = (bf16)W[k * (size_t)N + n];
}

// ---------------- layernorm (row of D), f32 in -> bf16 out -------------------
__global__ __launch_bounds__(256) void ln_kernel(const float* __restrict__ x,
                                                 const float* __restrict__ g,
                                                 const float* __restrict__ b,
                                                 bf16* __restrict__ out, int D) {
  int row = blockIdx.x;
  const float* xr = x + (size_t)row * D;
  float s = 0.f, s2 = 0.f;
  for (int i = threadIdx.x; i < D; i += 256) { float v = xr[i]; s += v; s2 += v * v; }
#pragma unroll
  for (int off = 1; off < 32; off <<= 1) {
    s += __shfl_xor(s, off, 32);
    s2 += __shfl_xor(s2, off, 32);
  }
  __shared__ float rs_[8], rs2_[8];
  __shared__ float mean_s, rstd_s;
  int w = threadIdx.x >> 5;
  if ((threadIdx.x & 31) == 0) { rs_[w] = s; rs2_[w] = s2; }
  __syncthreads();
  if (threadIdx.x == 0) {
    float S = 0.f, S2 = 0.f;
    for (int i = 0; i < 8; ++i) { S += rs_[i]; S2 += rs2_[i]; }
    float mean = S / D;
    float var = S2 / D - mean * mean;
    mean_s = mean;
    rstd_s = rsqrtf(var + 1e-5f);
  }
  __syncthreads();
  float mean = mean_s, rstd = rstd_s;
  bf16* orow = out + (size_t)row * D;
  for (int i = threadIdx.x; i < D; i += 256)
    orow[i] = (bf16)((xr[i] - mean) * rstd * g[i] + b[i]);
}

// -------- generic bf16 WMMA GEMM, 128x128 tile, 8 waves, async double-buffer --
enum { EP_F32 = 0, EP_RES_F32 = 1, EP_BF16 = 2, EP_GELU_BF16 = 3 };

template <int EP>
__global__ __launch_bounds__(256) void gemm_bf16(const bf16* __restrict__ A,   // [M][K]
                                                 const bf16* __restrict__ Bt,  // [N][K]
                                                 const float* __restrict__ bias,
                                                 const float* __restrict__ res,
                                                 float* __restrict__ outF,
                                                 bf16* __restrict__ outH,
                                                 int M, int N, int K) {
  __shared__ __align__(16) bf16 As[2][128 * 32];
  __shared__ __align__(16) bf16 Bs[2][128 * 32];
  int m0 = blockIdx.y * 128, n0 = blockIdx.x * 128;
  int tid = threadIdx.x;
  int w = tid >> 5, lane = tid & 31;
  int wm = w >> 2, wn = w & 3;            // 2x4 wave grid: 64 rows x 32 cols each
  int hf = lane >> 4, ln = lane & 15;
  v8f acc[4][2];
#pragma unroll
  for (int mt = 0; mt < 4; ++mt)
#pragma unroll
    for (int nt = 0; nt < 2; ++nt) acc[mt][nt] = zero8();

  // per-thread staging coordinates (2 x 16B per matrix per k-step)
  int sr0 = tid >> 2, sc0 = (tid & 3) * 8;          // e = tid
  int sr1 = (tid + 256) >> 2, sc1 = ((tid + 256) & 3) * 8;

  // prologue: stage tile 0 asynchronously
  async_cp16(&As[0][sr0 * 32 + sc0], &A[(size_t)(m0 + sr0) * K + sc0]);
  async_cp16(&Bs[0][sr0 * 32 + sc0], &Bt[(size_t)(n0 + sr0) * K + sc0]);
  async_cp16(&As[0][sr1 * 32 + sc1], &A[(size_t)(m0 + sr1) * K + sc1]);
  async_cp16(&Bs[0][sr1 * 32 + sc1], &Bt[(size_t)(n0 + sr1) * K + sc1]);
  async_wait0();
  __syncthreads();

  int nk = K >> 5;
  for (int kt = 0; kt < nk; ++kt) {
    int cur = kt & 1;
    if (kt + 1 < nk) {  // async prefetch next tile while computing this one
      int kk = (kt + 1) << 5;
      int nxt = cur ^ 1;
      async_cp16(&As[nxt][sr0 * 32 + sc0], &A[(size_t)(m0 + sr0) * K + kk + sc0]);
      async_cp16(&Bs[nxt][sr0 * 32 + sc0], &Bt[(size_t)(n0 + sr0) * K + kk + sc0]);
      async_cp16(&As[nxt][sr1 * 32 + sc1], &A[(size_t)(m0 + sr1) * K + kk + sc1]);
      async_cp16(&Bs[nxt][sr1 * 32 + sc1], &Bt[(size_t)(n0 + sr1) * K + kk + sc1]);
    }
    v16bf af[4], bfb[2];
#pragma unroll
    for (int mt = 0; mt < 4; ++mt) af[mt] = frag_A(&As[cur][(wm * 64 + mt * 16) * 32], 32);
#pragma unroll
    for (int nt = 0; nt < 2; ++nt) bfb[nt] = frag_B(&Bs[cur][(wn * 32 + nt * 16) * 32], 32);
#pragma unroll
    for (int mt = 0; mt < 4; ++mt)
#pragma unroll
      for (int nt = 0; nt < 2; ++nt)
        acc[mt][nt] = WMMA_BF16(af[mt], bfb[nt], acc[mt][nt]);
    async_wait0();
    __syncthreads();
  }

#pragma unroll
  for (int mt = 0; mt < 4; ++mt)
#pragma unroll
    for (int nt = 0; nt < 2; ++nt)
#pragma unroll
      for (int r = 0; r < 8; ++r) {
        int row = m0 + wm * 64 + mt * 16 + r + 8 * hf;
        int col = n0 + wn * 32 + nt * 16 + ln;
        float v = acc[mt][nt][r] + bias[col];
        size_t o = (size_t)row * N + col;
        if constexpr (EP == EP_F32) {
          outF[o] = v;
        } else if constexpr (EP == EP_RES_F32) {
          outF[o] = v + res[o];
        } else if constexpr (EP == EP_BF16) {
          outH[o] = (bf16)v;
        } else {  // exact GELU
          float gl = 0.5f * v * (1.f + erff(v * 0.70710678118f));
          outH[o] = (bf16)gl;
        }
      }
}

// ---------------- RoPE + pack to (B,H,T,hd) bf16; V transposed (B,H,hd,T) ----
__global__ __launch_bounds__(256) void pack_rope_kernel(const float* __restrict__ qkv,
                                                        const float* __restrict__ cosT,
                                                        const float* __restrict__ sinT,
                                                        bf16* __restrict__ Q,
                                                        bf16* __restrict__ Kd,
                                                        bf16* __restrict__ Vt, int T) {
  int idx = blockIdx.x * 256 + threadIdx.x;  // over B*T*H*32
  int i = idx & 31;
  int h = (idx >> 5) & 15;
  int bt = idx >> 9;
  int t = bt % T;
  int b = bt / T;
  const float* base = qkv + (size_t)bt * 3072;
  float c = cosT[t * 32 + i], s = sinT[t * 32 + i];
  float q1 = base[h * 64 + i], q2 = base[h * 64 + 32 + i];
  float k1 = base[1024 + h * 64 + i], k2 = base[1024 + h * 64 + 32 + i];
  float v1 = base[2048 + h * 64 + i], v2 = base[2048 + h * 64 + 32 + i];
  size_t bh = (size_t)(b * 16 + h);
  size_t qo = (bh * T + t) * 64;
  Q[qo + i] = (bf16)(q1 * c - q2 * s);
  Q[qo + 32 + i] = (bf16)(q1 * s + q2 * c);
  Kd[qo + i] = (bf16)(k1 * c - k2 * s);
  Kd[qo + 32 + i] = (bf16)(k1 * s + k2 * c);
  size_t vo = (bh * 64 + i) * T + t;
  Vt[vo] = (bf16)v1;
  Vt[vo + (size_t)32 * T] = (bf16)v2;
}

// ---------------- flash attention: 128 q rows/block, 64-key tiles ------------
__global__ __launch_bounds__(256) void attn_kernel(const bf16* __restrict__ Q,
                                                   const bf16* __restrict__ Kd,
                                                   const bf16* __restrict__ Vt,
                                                   bf16* __restrict__ O, int T) {
  __shared__ __align__(16) bf16 Qs[128 * 64];
  __shared__ __align__(16) bf16 Ks[64 * 64];
  __shared__ __align__(16) bf16 Vs[64 * 64];   // [hd][key]
  __shared__ __align__(16) bf16 Ps[128 * 64];  // per-wave 16x64 slices
  int bh = blockIdx.y;
  int q0 = blockIdx.x * 128;
  int tid = threadIdx.x, w = tid >> 5, lane = tid & 31;
  int hf = lane >> 4, ln = lane & 15;
  const bf16* Qg = Q + (size_t)bh * T * 64;
  const bf16* Kg = Kd + (size_t)bh * T * 64;
  const bf16* Vg = Vt + (size_t)bh * 64 * T;
#pragma unroll
  for (int i = 0; i < 4; ++i) {
    int e = tid + i * 256;
    int r = e >> 3, c = (e & 7) * 8;
    async_cp16(&Qs[r * 64 + c], &Qg[(size_t)(q0 + r) * 64 + c]);
  }
  async_wait0();
  __syncthreads();
  v16bf aq[2];
  aq[0] = frag_A(&Qs[(w * 16) * 64 + 0], 64);
  aq[1] = frag_A(&Qs[(w * 16) * 64 + 32], 64);
  float mrow[8], lrow[8];
  v8f acc[4];
#pragma unroll
  for (int r = 0; r < 8; ++r) { mrow[r] = -INFINITY; lrow[r] = 0.f; }
#pragma unroll
  for (int nt = 0; nt < 4; ++nt) acc[nt] = zero8();
  const float scale = 0.125f;  // 1/sqrt(64)
  int kb_end = (q0 + 128) / 64;
  for (int kb = 0; kb < kb_end; ++kb) {
    int k0 = kb * 64;
    __syncthreads();  // all waves done reading previous K/V tiles
#pragma unroll
    for (int i = 0; i < 2; ++i) {
      int e = tid + i * 256;
      int r = e >> 3, c = (e & 7) * 8;
      async_cp16(&Ks[r * 64 + c], &Kg[(size_t)(k0 + r) * 64 + c]);
      async_cp16(&Vs[r * 64 + c], &Vg[(size_t)r * T + k0 + c]);
    }
    async_wait0();
    __syncthreads();
    // S = Q K^T : hoist all K fragments, then burst 8 WMMAs
    v16bf bk[4][2];
#pragma unroll
    for (int nt = 0; nt < 4; ++nt)
#pragma unroll
      for (int ks = 0; ks < 2; ++ks)
        bk[nt][ks] = frag_B(&Ks[(nt * 16) * 64 + ks * 32], 64);
    v8f s[4];
#pragma unroll
    for (int nt = 0; nt < 4; ++nt) {
      s[nt] = WMMA_BF16(aq[0], bk[nt][0], zero8());
      s[nt] = WMMA_BF16(aq[1], bk[nt][1], s[nt]);
    }
    // causal mask + scale (C layout: row = r + 8*hf, col = ln)
#pragma unroll
    for (int nt = 0; nt < 4; ++nt)
#pragma unroll
      for (int r = 0; r < 8; ++r) {
        int col = k0 + nt * 16 + ln;
        int row = q0 + w * 16 + r + 8 * hf;
        float v = s[nt][r] * scale;
        s[nt][r] = (col <= row) ? v : -INFINITY;
      }
    // online softmax (rows live across 16 lanes within each half)
#pragma unroll
    for (int r = 0; r < 8; ++r) {
      float mx = fmaxf(fmaxf(s[0][r], s[1][r]), fmaxf(s[2][r], s[3][r]));
#pragma unroll
      for (int off = 1; off < 16; off <<= 1) mx = fmaxf(mx, __shfl_xor(mx, off, 16));
      float mnew = fmaxf(mrow[r], mx);
      float corr = __expf(mrow[r] - mnew);
      float lsum = 0.f;
#pragma unroll
      for (int nt = 0; nt < 4; ++nt) {
        float pv = __expf(s[nt][r] - mnew);
        s[nt][r] = pv;
        lsum += pv;
      }
#pragma unroll
      for (int off = 1; off < 16; off <<= 1) lsum += __shfl_xor(lsum, off, 16);
      lrow[r] = lrow[r] * corr + lsum;
      mrow[r] = mnew;
#pragma unroll
      for (int nt = 0; nt < 4; ++nt) acc[nt][r] *= corr;
    }
    // stage P (C-frag layout) -> LDS as A operand (wave-private region)
#pragma unroll
    for (int nt = 0; nt < 4; ++nt)
#pragma unroll
      for (int r = 0; r < 8; ++r)
        Ps[(w * 16 + r + 8 * hf) * 64 + nt * 16 + ln] = (bf16)s[nt][r];
    // O += P @ V : hoist P and V fragments, then burst 8 WMMAs
    v16bf pa[2], bv[4][2];
#pragma unroll
    for (int ks = 0; ks < 2; ++ks) pa[ks] = frag_A(&Ps[(w * 16) * 64 + ks * 32], 64);
#pragma unroll
    for (int nt = 0; nt < 4; ++nt)
#pragma unroll
      for (int ks = 0; ks < 2; ++ks)
        bv[nt][ks] = frag_B(&Vs[(nt * 16) * 64 + ks * 32], 64);
#pragma unroll
    for (int nt = 0; nt < 4; ++nt) {
      acc[nt] = WMMA_BF16(pa[0], bv[nt][0], acc[nt]);
      acc[nt] = WMMA_BF16(pa[1], bv[nt][1], acc[nt]);
    }
  }
  int b = bh >> 4, h = bh & 15;
#pragma unroll
  for (int nt = 0; nt < 4; ++nt)
#pragma unroll
    for (int r = 0; r < 8; ++r) {
      int row = q0 + w * 16 + r + 8 * hf;
      int col = h * 64 + nt * 16 + ln;
      float o = acc[nt][r] / lrow[r];
      O[((size_t)(b * T + row)) * 1024 + col] = (bf16)o;
    }
}

// -----------------------------------------------------------------------------
extern "C" void kernel_launch(void* const* d_in, const int* in_sizes, int n_in,
                              void* d_out, int out_size, void* d_ws, size_t ws_size,
                              hipStream_t stream) {
  (void)in_sizes; (void)n_in; (void)out_size; (void)ws_size;
  const int B = 2, T = 2048, D = 1024, H = 16;
  const int M = B * T;  // 4096
  const float* x        = (const float*)d_in[0];
  // d_in[1] = attn_mask (causal; computed analytically, not read)
  const float* rope_cos = (const float*)d_in[2];
  const float* rope_sin = (const float*)d_in[3];
  const float* ln1_g    = (const float*)d_in[4];
  const float* ln1_b    = (const float*)d_in[5];
  const float* qkv_w    = (const float*)d_in[6];
  const float* qkv_b    = (const float*)d_in[7];
  const float* out_w    = (const float*)d_in[8];
  const float* out_b    = (const float*)d_in[9];
  const float* ln2_g    = (const float*)d_in[10];
  const float* ln2_b    = (const float*)d_in[11];
  const float* mlp_w1   = (const float*)d_in[12];
  const float* mlp_b1   = (const float*)d_in[13];
  const float* mlp_w2   = (const float*)d_in[14];
  const float* mlp_b2   = (const float*)d_in[15];
  float* out = (float*)d_out;

  char* p = (char*)d_ws;
  auto take = [&](size_t bytes) {
    char* r = p;
    p += (bytes + 255) & ~(size_t)255;
    return r;
  };
  bf16* h1    = (bf16*)take((size_t)M * D * 2);
  bf16* qkvWT = (bf16*)take((size_t)3 * D * D * 2);
  bf16* outWT = (bf16*)take((size_t)D * D * 2);
  bf16* w1T   = (bf16*)take((size_t)4 * D * D * 2);
  bf16* w2T   = (bf16*)take((size_t)4 * D * D * 2);
  char* big   = take((size_t)M * 3 * D * 4);  // qkv f32; reused as MLP hidden a1
  float* qkvF = (float*)big;
  bf16* a1    = (bf16*)big;
  bf16* Qb    = (bf16*)take((size_t)M * D * 2);
  bf16* Kb    = (bf16*)take((size_t)M * D * 2);
  bf16* Vtb   = (bf16*)take((size_t)M * D * 2);
  bf16* attnO = (bf16*)take((size_t)M * D * 2);
  bf16* h2    = (bf16*)take((size_t)M * D * 2);

  // weights -> bf16, transposed to [N][K]
  wt_kernel<<<(3 * D * D + 255) / 256, 256, 0, stream>>>(qkv_w, qkvWT, D, 3 * D);
  wt_kernel<<<(D * D + 255) / 256, 256, 0, stream>>>(out_w, outWT, D, D);
  wt_kernel<<<(4 * D * D + 255) / 256, 256, 0, stream>>>(mlp_w1, w1T, D, 4 * D);
  wt_kernel<<<(4 * D * D + 255) / 256, 256, 0, stream>>>(mlp_w2, w2T, 4 * D, D);
  // LN1
  ln_kernel<<<M, 256, 0, stream>>>(x, ln1_g, ln1_b, h1, D);
  // QKV projection (f32 out for RoPE precision)
  gemm_bf16<EP_F32><<<dim3(3 * D / 128, M / 128), 256, 0, stream>>>(
      h1, qkvWT, qkv_b, nullptr, qkvF, nullptr, M, 3 * D, D);
  // RoPE + head pack (V transposed)
  pack_rope_kernel<<<(M * H * 32) / 256, 256, 0, stream>>>(qkvF, rope_cos, rope_sin,
                                                           Qb, Kb, Vtb, T);
  // causal flash attention
  attn_kernel<<<dim3(T / 128, B * H), 256, 0, stream>>>(Qb, Kb, Vtb, attnO, T);
  // out projection + residual -> d_out holds x1
  gemm_bf16<EP_RES_F32><<<dim3(D / 128, M / 128), 256, 0, stream>>>(
      attnO, outWT, out_b, x, out, nullptr, M, D, D);
  // LN2
  ln_kernel<<<M, 256, 0, stream>>>(out, ln2_g, ln2_b, h2, D);
  // MLP1 + exact GELU
  gemm_bf16<EP_GELU_BF16><<<dim3(4 * D / 128, M / 128), 256, 0, stream>>>(
      h2, w1T, mlp_b1, nullptr, nullptr, a1, M, 4 * D, D);
  // MLP2 + residual (in-place on d_out)
  gemm_bf16<EP_RES_F32><<<dim3(D / 128, M / 128), 256, 0, stream>>>(
      a1, w2T, mlp_b2, out, out, nullptr, M, D, 4 * D);
}
